// BertCRFNERModel_19473381720270
// MI455X (gfx1250) — compile-verified, matrix operand order checked by reading, not run on hardware
//
#include <hip/hip_runtime.h>

// ---------------- problem constants (match reference) ----------------
#define BB     8
#define SSEQ   256
#define HH     768
#define NLAYER 12
#define NHEAD  12
#define DHEAD  64
#define FFN    3072
#define NTOK   (BB*SSEQ)      // 2048
#define NBH    (BB*NHEAD)     // 96
#define NTAG   9
#define TSTART 7
#define TSTOP  8
#define TLEN   (SSEQ-2)       // 254
#define NEGV   (-10000.0f)

// GEMM epilogue modes
#define MODE_F32     0   // Cf[z*sCz + m*ldc + n] = val*scale + bias
#define MODE_GELU    1   // Cb[m*ldc + n] = bf16(gelu(val + bias))
#define MODE_QKHEAD  2   // Cb[((b*NH+h)*S + s)*DH + d] = bf16(val + bias)
#define MODE_VT      3   // Cb[((b*NH+h)*DH + d)*S + s] = bf16(val + bias)
#define MODE_CTX     4   // Cb[(z/NH)*S*H + m*H + (z%NH)*DH + n] = bf16(val)

// ---------------- CDNA5 async global->LDS path (guarded) ----------------
#if __has_builtin(__builtin_amdgcn_global_load_async_to_lds_b128) && \
    __has_builtin(__builtin_amdgcn_s_wait_asynccnt)
#define HAVE_ASYNC_LDS 1
typedef int v4i __attribute__((ext_vector_type(4)));
// pointer to int4 in global (AS1 / __device__) and LDS (AS3 / __shared__) spaces
typedef __attribute__((address_space(1))) v4i* gasp_t;
typedef __attribute__((address_space(3))) v4i* lasp_t;
// integer round-trip casts (low 32 bits of a generic LDS flat address are the
// LDS offset; AS3 pointers are 32-bit)
#define ASYNC_CP_B128(gp, lp)                                              \
  __builtin_amdgcn_global_load_async_to_lds_b128(                          \
      (gasp_t)(unsigned long long)(size_t)(gp),                            \
      (lasp_t)(unsigned int)(size_t)(lp), 0, 0)
#else
#define HAVE_ASYNC_LDS 0
#endif

// ---------------- WMMA types ----------------
typedef __attribute__((ext_vector_type(16))) __bf16 v16bf;
typedef __attribute__((ext_vector_type(8)))  __bf16 bh8;
typedef __attribute__((ext_vector_type(8)))  float  v8f;

__device__ __forceinline__ __bf16 to_bf16(float f){
  unsigned int u = __float_as_uint(f);
  u += 0x7fffu + ((u >> 16) & 1u);          // round-to-nearest-even
  unsigned short r = (unsigned short)(u >> 16);
  return __builtin_bit_cast(__bf16, r);
}

// Load one 16x32 bf16 WMMA fragment row for this lane.
// 16-bit A 16x32 (ISA): lane = m (mod 16), lane-half selects k offset 8;
// elements 0..7  -> k = half*8 + 0..7   (contiguous 16B)
// elements 8..15 -> k = 16 + half*8 + 0..7 (contiguous 16B)
// B operands are stored N-major so the same loader serves both.
__device__ __forceinline__ v16bf load_frag(const __bf16* rowk0, int half){
  const bh8* p = (const bh8*)(rowk0 + half*8);
  bh8 lo = p[0];
  bh8 hi = p[2];   // rowk0 + half*8 + 16
  v16bf r;
#pragma unroll
  for(int i=0;i<8;i++){ r[i]=lo[i]; r[8+i]=hi[i]; }
  return r;
}

// ---------------- fp32 -> bf16 transpose-convert: W[K x N] -> Wt[N x K] ----
__global__ __launch_bounds__(256) void convt_kernel(const float* __restrict__ W,
                                                    __bf16* __restrict__ Wt,
                                                    int K, int N){
  __shared__ float t[32][33];
  int nt = blockIdx.x*32, kt = blockIdx.y*32;
  int tx = threadIdx.x, ty = threadIdx.y;     // block (32,8)
#pragma unroll
  for(int i=0;i<4;i++)
    t[ty+i*8][tx] = W[(size_t)(kt+ty+i*8)*N + nt + tx];
  __syncthreads();
#pragma unroll
  for(int i=0;i<4;i++)
    Wt[(size_t)(nt+ty+i*8)*K + kt + tx] = to_bf16(t[tx][ty+i*8]);
}

// ---------------- batched WMMA GEMM ----------------
// C[z][MxN] = A[z][MxK](lda) * Bt[z][NxK](ldb) ; double-buffered LDS staging.
#define BM 128
#define BN 64
#define BK 32
__global__ __launch_bounds__(256) void gemm_kernel(
    const __bf16* __restrict__ A,  unsigned long long sAz, int lda,
    const __bf16* __restrict__ Bt, unsigned long long sBz, int ldb,
    const float*  __restrict__ bias,
    float*  __restrict__ Cf, __bf16* __restrict__ Cb,
    int M, int N, int K, int ldc, unsigned long long sCz,
    float scale, int mode){
  __shared__ __align__(16) __bf16 sA[2][BM*BK];
  __shared__ __align__(16) __bf16 sB[2][BN*BK];
  int tid  = threadIdx.x;
  int wave = tid >> 5, lane = tid & 31;
  int half = lane >> 4, lr = lane & 15;
  int wm = wave & 3, wn = wave >> 2;            // 4x2 wave grid -> 128x64 tile
  int m0 = blockIdx.y*BM, n0 = blockIdx.x*BN;
  int z  = blockIdx.z;
  const __bf16* Az = A  + (size_t)z*sAz;
  const __bf16* Bz = Bt + (size_t)z*sBz;

  // per-thread staging coordinates (b128 chunks)
  int ar0 = tid >> 2,        ac0 = (tid & 3)*8;          // A rows 0..63
  int ar1 = (tid>>2) + 64,   ac1 = ac0;                  // A rows 64..127
  int br  = tid >> 2,        bc  = ac0;                  // B rows 0..63
  const __bf16* gA0 = Az + (size_t)(m0+ar0)*lda + ac0;
  const __bf16* gA1 = Az + (size_t)(m0+ar1)*lda + ac1;
  const __bf16* gB0 = Bz + (size_t)(n0+br )*ldb + bc;

  v8f acc[2][2];
#pragma unroll
  for(int i=0;i<2;i++)
#pragma unroll
    for(int j=0;j<2;j++)
#pragma unroll
      for(int e=0;e<8;e++) acc[i][j][e] = 0.0f;

  int nk = K/BK;
  // preload tile 0
#if HAVE_ASYNC_LDS
  ASYNC_CP_B128(gA0, sA[0] + ar0*BK + ac0);
  ASYNC_CP_B128(gA1, sA[0] + ar1*BK + ac1);
  ASYNC_CP_B128(gB0, sB[0] + br *BK + bc );
  __builtin_amdgcn_s_wait_asynccnt(0);
#else
  {
    bh8 a0 = *(const bh8*)gA0;
    bh8 a1 = *(const bh8*)gA1;
    bh8 b0 = *(const bh8*)gB0;
    *(bh8*)(sA[0] + ar0*BK + ac0) = a0;
    *(bh8*)(sA[0] + ar1*BK + ac1) = a1;
    *(bh8*)(sB[0] + br *BK + bc ) = b0;
  }
#endif
  __syncthreads();

  int cur = 0;
  for(int it=0; it<nk; ++it){
    bool has_next = (it+1 < nk);
    int kn = (it+1)*BK;
#if HAVE_ASYNC_LDS
    if(has_next){
      int nxt = cur ^ 1;
      ASYNC_CP_B128(gA0 + kn, sA[nxt] + ar0*BK + ac0);
      ASYNC_CP_B128(gA1 + kn, sA[nxt] + ar1*BK + ac1);
      ASYNC_CP_B128(gB0 + kn, sB[nxt] + br *BK + bc );
    }
#else
    bh8 ga0, ga1, gb0;
    if(has_next){
      ga0 = *(const bh8*)(gA0 + kn);
      ga1 = *(const bh8*)(gA1 + kn);
      gb0 = *(const bh8*)(gB0 + kn);
    }
#endif
    if(it+2 < nk){
      int kp = (it+2)*BK;
      __builtin_prefetch(gA0 + kp, 0, 1);
      __builtin_prefetch(gB0 + kp, 0, 1);
    }

    v16bf af[2], bfr[2];
#pragma unroll
    for(int i=0;i<2;i++)
      af[i]  = load_frag(sA[cur] + (wm*32 + i*16 + lr)*BK, half);
#pragma unroll
    for(int j=0;j<2;j++)
      bfr[j] = load_frag(sB[cur] + (wn*32 + j*16 + lr)*BK, half);

#pragma unroll
    for(int i=0;i<2;i++)
#pragma unroll
      for(int j=0;j<2;j++)
        acc[i][j] = __builtin_amdgcn_wmma_f32_16x16x32_bf16(
            false, af[i], false, bfr[j], (short)0, acc[i][j], false, false);

    if(has_next){
#if HAVE_ASYNC_LDS
      __builtin_amdgcn_s_wait_asynccnt(0);
#else
      int nxt = cur ^ 1;
      *(bh8*)(sA[nxt] + ar0*BK + ac0) = ga0;
      *(bh8*)(sA[nxt] + ar1*BK + ac1) = ga1;
      *(bh8*)(sB[nxt] + br *BK + bc ) = gb0;
#endif
      __syncthreads();
      cur ^= 1;
    }
  }

  // epilogue: acc VGPR r -> m = r + half*8 (within 16), n = lane&15
  size_t ctxbase = 0;
  if(mode == MODE_CTX)
    ctxbase = (size_t)(z/NHEAD)*SSEQ*HH + (size_t)(z%NHEAD)*DHEAD;
#pragma unroll
  for(int i=0;i<2;i++){
#pragma unroll
    for(int j=0;j<2;j++){
#pragma unroll
      for(int r=0;r<8;r++){
        int m = m0 + wm*32 + i*16 + half*8 + r;
        int n = n0 + wn*32 + j*16 + lr;
        float val = acc[i][j][r]*scale;
        if(bias) val += bias[n];
        if(mode == MODE_F32){
          Cf[(size_t)z*sCz + (size_t)m*ldc + n] = val;
        }else if(mode == MODE_GELU){
          float g = 0.5f*val*(1.0f + erff(val*0.70710678118654752440f));
          Cb[(size_t)m*ldc + n] = to_bf16(g);
        }else if(mode == MODE_QKHEAD){
          int b = m >> 8, s = m & (SSEQ-1), h = n >> 6, d = n & 63;
          Cb[((size_t)(b*NHEAD+h)*SSEQ + s)*DHEAD + d] = to_bf16(val);
        }else if(mode == MODE_VT){
          int b = m >> 8, s = m & (SSEQ-1), h = n >> 6, d = n & 63;
          Cb[((size_t)(b*NHEAD+h)*DHEAD + d)*SSEQ + s] = to_bf16(val);
        }else{ // MODE_CTX
          Cb[ctxbase + (size_t)m*HH + n] = to_bf16(val);
        }
      }
    }
  }
}

// ---------------- embedding sum ----------------
__global__ __launch_bounds__(256) void embed_kernel(const int* __restrict__ sent,
    const float* __restrict__ wemb, const float* __restrict__ pemb,
    const float* __restrict__ temb, float* __restrict__ x){
  int tok = blockIdx.x;
  int sidx = tok & (SSEQ-1);
  int id = sent[tok];
  for(int hh=threadIdx.x; hh<HH; hh+=256)
    x[(size_t)tok*HH + hh] = wemb[(size_t)id*HH + hh] + pemb[(size_t)sidx*HH + hh] + temb[hh];
}

// ---------------- (optional add) + LayerNorm; writes fp32 + bf16 ----------
__global__ __launch_bounds__(256) void add_ln_kernel(const float* __restrict__ xin,
    const float* __restrict__ add, const float* __restrict__ gam,
    const float* __restrict__ bet, float* __restrict__ xout,
    __bf16* __restrict__ xbout){
  __shared__ float red[256];
  int tok = blockIdx.x, tid = threadIdx.x;
  float vals[3];
  float s = 0.f;
#pragma unroll
  for(int i=0;i<3;i++){
    int hh = tid + i*256;
    float vv = xin[(size_t)tok*HH + hh];
    if(add) vv += add[(size_t)tok*HH + hh];
    vals[i] = vv; s += vv;
  }
  red[tid] = s; __syncthreads();
  for(int off=128; off>0; off>>=1){ if(tid<off) red[tid] += red[tid+off]; __syncthreads(); }
  float mean = red[0]*(1.0f/HH);
  __syncthreads();
  float s2 = 0.f;
#pragma unroll
  for(int i=0;i<3;i++){ float d = vals[i]-mean; s2 += d*d; }
  red[tid] = s2; __syncthreads();
  for(int off=128; off>0; off>>=1){ if(tid<off) red[tid] += red[tid+off]; __syncthreads(); }
  float inv = rsqrtf(red[0]*(1.0f/HH) + 1e-12f);
#pragma unroll
  for(int i=0;i<3;i++){
    int hh = tid + i*256;
    float y = (vals[i]-mean)*inv*gam[hh] + bet[hh];
    xout[(size_t)tok*HH + hh]  = y;
    xbout[(size_t)tok*HH + hh] = to_bf16(y);
  }
}

// ---------------- softmax row (fp32 in, bf16 out) ----------------
__global__ __launch_bounds__(256) void softmax_kernel(const float* __restrict__ att,
    __bf16* __restrict__ attb){
  __shared__ float red[256];
  int row = blockIdx.x;            // bh*S + q
  int tid = threadIdx.x;
  float d = att[(size_t)row*SSEQ + tid];   // already scaled by 1/sqrt(dh)
  red[tid] = d; __syncthreads();
  for(int off=128; off>0; off>>=1){ if(tid<off) red[tid] = fmaxf(red[tid], red[tid+off]); __syncthreads(); }
  float mx = red[0]; __syncthreads();
  float ex = __expf(d - mx);
  red[tid] = ex; __syncthreads();
  for(int off=128; off>0; off>>=1){ if(tid<off) red[tid] += red[tid+off]; __syncthreads(); }
  attb[(size_t)row*SSEQ + tid] = to_bf16(ex / red[0]);
}

// ---------------- final FC: feats = x @ fc_w + fc_b (N=9) ----------------
__global__ __launch_bounds__(256) void fc_kernel(const float* __restrict__ x,
    const float* __restrict__ w, const float* __restrict__ bias,
    float* __restrict__ feats){
  __shared__ float red[NTAG*256];
  int tok = blockIdx.x, tid = threadIdx.x;
  float p[NTAG];
#pragma unroll
  for(int t=0;t<NTAG;t++) p[t] = 0.f;
  for(int hh=tid; hh<HH; hh+=256){
    float xv = x[(size_t)tok*HH + hh];
#pragma unroll
    for(int t=0;t<NTAG;t++) p[t] += xv*w[hh*NTAG + t];
  }
#pragma unroll
  for(int t=0;t<NTAG;t++) red[t*256 + tid] = p[t];
  __syncthreads();
  for(int off=128; off>0; off>>=1){
    if(tid<off){
#pragma unroll
      for(int t=0;t<NTAG;t++) red[t*256 + tid] += red[t*256 + tid + off];
    }
    __syncthreads();
  }
  if(tid < NTAG) feats[(size_t)tok*NTAG + tid] = red[tid*256] + bias[tid];
}

// ---------------- Viterbi decode (first-max argmax like jnp.argmax) --------
__global__ __launch_bounds__(128) void viterbi_kernel(const float* __restrict__ feats,
    const float* __restrict__ tr, int* __restrict__ bps, float* __restrict__ out){
  __shared__ float fv[BB][NTAG];
  __shared__ float nv[BB][NTAG];
  int tid = threadIdx.x;
  int b = tid >> 4, i = tid & 15;
  bool act = (b < BB) && (i < NTAG);
  if(act) fv[b][i] = (i == TSTART) ? 0.f : NEGV;
  __syncthreads();
  for(int t=0; t<TLEN; t++){
    if(act){
      float best = -3.4e38f; int bj = 0;
#pragma unroll
      for(int j=0;j<NTAG;j++){
        float vv = fv[b][j] + tr[i*NTAG + j];
        if(vv > best){ best = vv; bj = j; }
      }
      nv[b][i] = best + feats[((size_t)(b*SSEQ) + (t+1))*NTAG + i];
      bps[((size_t)t*BB + b)*NTAG + i] = bj;
    }
    __syncthreads();
    if(act) fv[b][i] = nv[b][i];
    __syncthreads();
  }
  if(act && i==0){
    float best = -3.4e38f; int bj = 0;
#pragma unroll
    for(int j=0;j<NTAG;j++){
      float vv = fv[b][j] + tr[TSTOP*NTAG + j];
      if(vv > best){ best = vv; bj = j; }
    }
    out[b] = best;
    int tag = bj;
    for(int t=TLEN-1; t>=0; t--){
      out[BB + (size_t)b*TLEN + t] = (float)tag;
      tag = bps[((size_t)t*BB + b)*NTAG + tag];
    }
  }
}

// ---------------- host orchestration ----------------
extern "C" void kernel_launch(void* const* d_in, const int* in_sizes, int n_in,
                              void* d_out, int out_size, void* d_ws, size_t ws_size,
                              hipStream_t stream){
  (void)in_sizes; (void)n_in; (void)out_size; (void)ws_size;
  const int*   sent = (const int*)  d_in[0];
  const float* wemb = (const float*)d_in[2];
  const float* pemb = (const float*)d_in[3];
  const float* temb = (const float*)d_in[4];
  const float* elns = (const float*)d_in[5];
  const float* elnb = (const float*)d_in[6];
  const float* Wq=(const float*)d_in[7];   const float* bq=(const float*)d_in[8];
  const float* Wk=(const float*)d_in[9];   const float* bk=(const float*)d_in[10];
  const float* Wv=(const float*)d_in[11];  const float* bv=(const float*)d_in[12];
  const float* Wo=(const float*)d_in[13];  const float* bo=(const float*)d_in[14];
  const float* l1s=(const float*)d_in[15]; const float* l1b=(const float*)d_in[16];
  const float* W1=(const float*)d_in[17];  const float* b1=(const float*)d_in[18];
  const float* W2=(const float*)d_in[19];  const float* b2=(const float*)d_in[20];
  const float* l2s=(const float*)d_in[21]; const float* l2b=(const float*)d_in[22];
  const float* fcw=(const float*)d_in[23]; const float* fcb=(const float*)d_in[24];
  const float* trans=(const float*)d_in[25];
  float* out = (float*)d_out;

  char* cur = (char*)d_ws;
  auto carve = [&](size_t bytes)->char*{
    char* r = cur; cur += (bytes + 255) & ~(size_t)255; return r;
  };
  float*  x    = (float*) carve((size_t)NTOK*HH*4);
  __bf16* xb   = (__bf16*)carve((size_t)NTOK*HH*2);
  __bf16* qh   = (__bf16*)carve((size_t)NBH*SSEQ*DHEAD*2);   // [bh][s][d]
  __bf16* kh   = (__bf16*)carve((size_t)NBH*SSEQ*DHEAD*2);   // [bh][s][d]
  __bf16* vt   = (__bf16*)carve((size_t)NBH*DHEAD*SSEQ*2);   // [bh][d][s]
  float*  att  = (float*) carve((size_t)NBH*SSEQ*SSEQ*4);    // fp32 scores
  __bf16* attb = (__bf16*)carve((size_t)NBH*SSEQ*SSEQ*2);    // bf16 probs
  __bf16* ctxb = (__bf16*)carve((size_t)NTOK*HH*2);
  float*  obuf = (float*) carve((size_t)NTOK*HH*4);
  __bf16* hbuf = (__bf16*)carve((size_t)NTOK*FFN*2);
  float*  feats= (float*) carve((size_t)NTOK*NTAG*4);
  int*    bps  = (int*)   carve((size_t)TLEN*BB*NTAG*4);
  __bf16* wqt  = (__bf16*)carve((size_t)HH*HH*2);
  __bf16* wkt  = (__bf16*)carve((size_t)HH*HH*2);
  __bf16* wvt  = (__bf16*)carve((size_t)HH*HH*2);
  __bf16* wot  = (__bf16*)carve((size_t)HH*HH*2);
  __bf16* w1t  = (__bf16*)carve((size_t)FFN*HH*2);
  __bf16* w2t  = (__bf16*)carve((size_t)HH*FFN*2);

  embed_kernel<<<NTOK,256,0,stream>>>(sent, wemb, pemb, temb, x);
  add_ln_kernel<<<NTOK,256,0,stream>>>(x, nullptr, elns, elnb, x, xb);

  dim3 cblk(32,8);
  dim3 gH  (HH/BN,   NTOK/BM, 1);     // 2048 x 768
  dim3 gFF (FFN/BN,  NTOK/BM, 1);     // 2048 x 3072
  dim3 gQK (SSEQ/BN, SSEQ/BM, NBH);   // 256 x 256 per (b,h)
  dim3 gCTX(DHEAD/BN,SSEQ/BM, NBH);   // 256 x 64  per (b,h)
  const unsigned long long sHD = (unsigned long long)SSEQ*DHEAD;   // 16384
  const unsigned long long sSS = (unsigned long long)SSEQ*SSEQ;    // 65536

  for(int l=0; l<NLAYER; l++){
    convt_kernel<<<dim3(HH/32, HH/32), cblk,0,stream>>>(Wq+(size_t)l*HH*HH, wqt, HH, HH);
    convt_kernel<<<dim3(HH/32, HH/32), cblk,0,stream>>>(Wk+(size_t)l*HH*HH, wkt, HH, HH);
    convt_kernel<<<dim3(HH/32, HH/32), cblk,0,stream>>>(Wv+(size_t)l*HH*HH, wvt, HH, HH);
    convt_kernel<<<dim3(HH/32, HH/32), cblk,0,stream>>>(Wo+(size_t)l*HH*HH, wot, HH, HH);
    convt_kernel<<<dim3(FFN/32, HH/32), cblk,0,stream>>>(W1+(size_t)l*HH*FFN, w1t, HH, FFN);
    convt_kernel<<<dim3(HH/32, FFN/32), cblk,0,stream>>>(W2+(size_t)l*FFN*HH, w2t, FFN, HH);

    // QKV projections (WMMA), scattered into attention-friendly layouts
    gemm_kernel<<<gH,256,0,stream>>>(xb,0,HH, wqt,0,HH, bq+(size_t)l*HH,
        nullptr, qh, NTOK,HH,HH, 0,0, 1.f, MODE_QKHEAD);
    gemm_kernel<<<gH,256,0,stream>>>(xb,0,HH, wkt,0,HH, bk+(size_t)l*HH,
        nullptr, kh, NTOK,HH,HH, 0,0, 1.f, MODE_QKHEAD);
    gemm_kernel<<<gH,256,0,stream>>>(xb,0,HH, wvt,0,HH, bv+(size_t)l*HH,
        nullptr, vt, NTOK,HH,HH, 0,0, 1.f, MODE_VT);

    // scores = Q K^T / 8  (batched WMMA over 96 heads)
    gemm_kernel<<<gQK,256,0,stream>>>(qh,sHD,DHEAD, kh,sHD,DHEAD, nullptr,
        att, nullptr, SSEQ,SSEQ,DHEAD, SSEQ,sSS, 0.125f, MODE_F32);
    softmax_kernel<<<NBH*SSEQ,256,0,stream>>>(att, attb);
    // ctx = att V  (batched WMMA), scatter back to token-major bf16
    gemm_kernel<<<gCTX,256,0,stream>>>(attb,sSS,SSEQ, vt,sHD,SSEQ, nullptr,
        nullptr, ctxb, SSEQ,DHEAD,SSEQ, 0,0, 1.f, MODE_CTX);

    gemm_kernel<<<gH,256,0,stream>>>(ctxb,0,HH, wot,0,HH, bo+(size_t)l*HH,
        obuf, nullptr, NTOK,HH,HH, HH,0, 1.f, MODE_F32);
    add_ln_kernel<<<NTOK,256,0,stream>>>(x, obuf, l1s+(size_t)l*HH, l1b+(size_t)l*HH, x, xb);

    gemm_kernel<<<gFF,256,0,stream>>>(xb,0,HH, w1t,0,HH, b1+(size_t)l*FFN,
        nullptr, hbuf, NTOK,FFN,HH, FFN,0, 1.f, MODE_GELU);
    gemm_kernel<<<gH ,256,0,stream>>>(hbuf,0,FFN, w2t,0,FFN, b2+(size_t)l*HH,
        obuf, nullptr, NTOK,HH,FFN, HH,0, 1.f, MODE_F32);
    add_ln_kernel<<<NTOK,256,0,stream>>>(x, obuf, l2s+(size_t)l*HH, l2b+(size_t)l*HH, x, xb);
  }

  fc_kernel<<<NTOK,256,0,stream>>>(x, fcw, fcb, feats);
  viterbi_kernel<<<1,128,0,stream>>>(feats, trans, bps, out);
}
